// BaseDefender_37314675867667
// MI455X (gfx1250) — compile-verified, matrix operand order checked by reading, not run on hardware
//
#include <hip/hip_runtime.h>
#include <stdint.h>

typedef __attribute__((ext_vector_type(16))) _Float16 v16h;
typedef __attribute__((ext_vector_type(8)))  _Float16 v8h;
typedef __attribute__((ext_vector_type(8)))  float    v8f;

#define NS 8192
#define NT 8192
#define CC 91
#define CP 96          // pad 91 -> 96 = 3 * K32 WMMA chunks; row stride 192B (16B-mult)
#define NO_OBJ 90
#define THR 0.75f
#define TCHUNK 1024    // teacher boxes staged per LDS pass (16KB)

// ---------------------------------------------------------------------------
// Workspace: [0 .. NS) packed u64 (iou_bits<<32 | ~argmax), then 4 floats:
// accum[0]=sum above*kl, accum[1]=n_above, accum[2]=sum below nll, accum[3]=n_below
// ---------------------------------------------------------------------------

__global__ void init_ws(unsigned long long* packed, float* accum) {
  int i = blockIdx.x * 256 + threadIdx.x;
  if (i < NS) packed[i] = 0ull;
  if (i < 4)  accum[i] = 0.f;
}

// ---------------------------------------------------------------------------
// Phase 1: per-student max IoU + argmax over a 1024-teacher chunk.
// Teacher boxes staged to LDS with gfx1250 async copy (ASYNCcnt path).
// Candidate comparison without division: inter/den > binter/bden
// <=> inter*bden > binter*den (all positive). One divide per chunk.
// Cross-chunk combine: atomic max on packed (iou_bits, ~idx): float order
// preserved, ties resolve to smallest teacher index (argmax semantics).
// ---------------------------------------------------------------------------
__global__ __launch_bounds__(256) void iou_kernel(
    const float4* __restrict__ bs, const float* __restrict__ bt,
    unsigned long long* __restrict__ packed) {
  __shared__ __align__(16) float4 shT[TCHUNK];
  __shared__ float  shAr[TCHUNK];

  const int tid    = threadIdx.x;
  const int s      = blockIdx.x * 256 + tid;
  const int chunk0 = blockIdx.y * TCHUNK;

  // Async-stage 1024 teacher boxes (16B each) into LDS: GVS mode,
  // saddr = uniform base, vaddr = per-lane byte offset, vdst = LDS byte addr.
  const float* base = bt + (size_t)chunk0 * 4;
  for (int j = 0; j < TCHUNK / 256; ++j) {
    int r = tid + j * 256;
    unsigned ldsoff = (unsigned)(uintptr_t)&shT[r];
    unsigned voff   = (unsigned)(r * 16);
    asm volatile("global_load_async_to_lds_b128 %0, %1, %2"
                 :: "v"(ldsoff), "v"(voff), "s"(base) : "memory");
  }
  asm volatile("s_wait_asynccnt 0" ::: "memory");
  __syncthreads();

  // Precompute teacher areas once per chunk.
  for (int j = tid; j < TCHUNK; j += 256) {
    float4 t = shT[j];
    shAr[j] = (t.z - t.x + 1.f) * (t.w - t.y + 1.f);
  }
  __syncthreads();

  float4 sb = bs[s];
  const float sx1 = sb.x, sy1 = sb.y, sx2 = sb.z, sy2 = sb.w;
  const float as = (sx2 - sx1 + 1.f) * (sy2 - sy1 + 1.f);

  float binter = 0.f, bden = 1.f;
  int   bidx = chunk0;
#pragma unroll 4
  for (int j = 0; j < TCHUNK; ++j) {
    float4 t = shT[j];                       // LDS broadcast across the wave
    float iw = fminf(sx2, t.z) - fmaxf(sx1, t.x) + 1.f;
    float ih = fminf(sy2, t.w) - fmaxf(sy1, t.y) + 1.f;
    iw = fmaxf(iw, 0.f); ih = fmaxf(ih, 0.f);
    float inter = iw * ih;
    float den   = as + shAr[j] - inter;
    if (inter * bden > binter * den) {       // strict > keeps earliest index
      binter = inter; bden = den; bidx = chunk0 + j;
    }
  }
  float iou = binter / bden;                 // one divide per chunk
  unsigned long long p =
      ((unsigned long long)__float_as_uint(iou) << 32) |
      (unsigned long long)(~(unsigned)bidx);
  atomicMax(&packed[s], p);
}

// ---------------------------------------------------------------------------
// Phase 2: one wave per 16 students. WMMA computes the 16x16 product of
// A = masked gathered teacher probs (16 x 96, f16) times B^T = log softmax
// of student (96 x 16, f16); only the diagonal (per-row dot products) is
// kept. Sum pt*log(pt) is accumulated via LDS ds_add_f32 while staging.
//
// Operand marshalling: 16-bit A layout (ISA 7.12.2) per lane is two
// contiguous 8-half runs: K in [off,off+8) and [off+16,off+24), off = 0
// (lanes 0-15) or 8 (lanes 16-31). With 16B-aligned rows of 192B stride,
// each operand chunk is two ds_load_b128.
// ---------------------------------------------------------------------------
__global__ __launch_bounds__(32) void kl_kernel(
    const unsigned long long* __restrict__ packed,
    const float* __restrict__ ps, const float* __restrict__ pt,
    float* accum) {
  __shared__ __align__(16) _Float16 shA[16][CP];  // gathered pt rows (0 if below/pad)
  __shared__ __align__(16) _Float16 shB[16][CP];  // log(ps) rows
  __shared__ float    shRow[16];                  // sum pt*log(pt) per above row
  __shared__ int      shIdx[16];
  __shared__ float    shAb[16];

  const int lane = threadIdx.x;
  const int s0   = blockIdx.x * 16;

  if (lane < 16) {
    unsigned long long p = packed[s0 + lane];
    float miou = __uint_as_float((unsigned)(p >> 32));
    int   idx  = (int)(~(unsigned)(p & 0xffffffffull));
    float ab   = (miou > THR) ? 1.f : 0.f;
    shIdx[lane] = idx; shAb[lane] = ab; shRow[lane] = 0.f;
    if (ab == 0.f) {
      atomicAdd(&accum[2], -__logf(ps[(size_t)(s0 + lane) * CC + NO_OBJ]));
      atomicAdd(&accum[3], 1.f);
    } else {
      atomicAdd(&accum[1], 1.f);
    }
  }
  __syncthreads();

  for (int e = lane; e < 16 * CP; e += 32) {
    int r = e / CP, c = e % CP;
    float pv = 0.f;
    if (c < CC && shAb[r] > 0.f) pv = pt[(size_t)shIdx[r] * CC + c];
    shA[r][c] = (_Float16)pv;
    if (pv > 0.f) atomicAdd(&shRow[r], pv * __logf(pv));   // ds_add_f32
    float lv = (c < CC) ? __logf(ps[(size_t)(s0 + r) * CC + c]) : 0.f;
    shB[r][c] = (_Float16)lv;
  }
  __syncthreads();

  v8f acc = {};
  const int m   = lane & 15;
  const int off = (lane >> 4) << 3;        // 0 for lanes 0-15, 8 for 16-31
#pragma unroll
  for (int k0 = 0; k0 < CP; k0 += 32) {
    const v8h* pa = (const v8h*)&shA[m][k0 + off];
    const v8h* pb = (const v8h*)&shB[m][k0 + off];
    v8h a_lo = pa[0];                      // halves 0-7  -> K off..off+7
    v8h a_hi = pa[2];                      // halves 8-15 -> K off+16..off+23
    v8h b_lo = pb[0];
    v8h b_hi = pb[2];
    v16h a = __builtin_shufflevector(a_lo, a_hi,
        0, 1, 2, 3, 4, 5, 6, 7, 8, 9, 10, 11, 12, 13, 14, 15);
    v16h b = __builtin_shufflevector(b_lo, b_hi,
        0, 1, 2, 3, 4, 5, 6, 7, 8, 9, 10, 11, 12, 13, 14, 15);
    acc = __builtin_amdgcn_wmma_f32_16x16x32_f16(
        /*neg_a=*/false, a, /*neg_b=*/false, b,
        /*c_mod=*/(short)0, acc, /*reuse_a=*/false, /*reuse_b=*/false);
  }

  // Diagonal of 16x16 f32 D: lanes 0-7 -> acc[lane] (M=N=lane);
  // lanes 24-31 -> acc[lane-24] (M=N=lane-16).
  float d = 0.f;
  if (lane < 8)        d = acc[lane];
  else if (lane >= 24) d = acc[lane - 24];

  // above*kl contribution for this group: sum(shRow) - sum(diag)
  float v = ((lane < 16) ? shRow[lane] : 0.f) - d;
  for (int o = 16; o > 0; o >>= 1) v += __shfl_xor(v, o, 32);
  if (lane == 0) atomicAdd(&accum[0], v);
}

__global__ void finalize(const float* accum, float* out) {
  float sA = accum[0], nA = accum[1], sB = accum[2], nB = accum[3];
  float r = 0.f;
  if (nA > 0.f) r += sA / (nA * (float)CC);
  if (nB > 0.f) r += sB / (nB * (float)CC);
  out[0] = r;
}

extern "C" void kernel_launch(void* const* d_in, const int* in_sizes, int n_in,
                              void* d_out, int out_size, void* d_ws, size_t ws_size,
                              hipStream_t stream) {
  const float4* bs = (const float4*)d_in[0];
  const float*  bt = (const float*)d_in[1];
  const float*  ps = (const float*)d_in[2];
  const float*  pt = (const float*)d_in[3];

  unsigned long long* packed = (unsigned long long*)d_ws;
  float* accum = (float*)((char*)d_ws + (size_t)NS * sizeof(unsigned long long));

  init_ws<<<(NS + 255) / 256, 256, 0, stream>>>(packed, accum);
  dim3 g(NS / 256, NT / TCHUNK);
  iou_kernel<<<g, 256, 0, stream>>>(bs, bt, packed);
  kl_kernel<<<NS / 16, 32, 0, stream>>>(packed, ps, pt, accum);
  finalize<<<1, 1, 0, stream>>>(accum, (float*)d_out);
}
